// GEARS_Model_9225589751886
// MI455X (gfx1250) — matrix operand order repaired and checked by reading, not compiled, not run on hardware
//
#include <hip/hip_runtime.h>
#include <hip/hip_bf16.h>

// ---------------- CDNA5 WMMA types ----------------
typedef __attribute__((ext_vector_type(16))) __bf16 v16bf;
typedef __attribute__((ext_vector_type(8)))  float  v8f;

// Model constants
static constexpr int cB   = 32;
static constexpr int cG   = 5000;
static constexpr int cP   = 5000;
static constexpr int cH   = 64;
static constexpr int cN   = cB * cG;       // 160000
static constexpr int cECO = 1600000;
static constexpr int cEGG = 100000;

// ---------------- small utility kernels ----------------
__global__ void k_fill(float* p, float v, int n) {
  int i = blockIdx.x * blockDim.x + threadIdx.x;
  if (i < n) p[i] = v;
}

// rows renormed to L2 norm <= 1 (nn.Embedding max_norm)
__global__ void k_renorm(const float* __restrict__ X, float* __restrict__ Y, int rows) {
  int r = blockIdx.x * blockDim.x + threadIdx.x;
  if (r >= rows) return;
  const float* x = X + (size_t)r * cH;
  float ss = 0.f;
  #pragma unroll
  for (int h = 0; h < cH; ++h) ss += x[h] * x[h];
  float n = sqrtf(ss);
  float sc = fminf(1.f, 1.f / fmaxf(n, 1e-12f));
  float* y = Y + (size_t)r * cH;
  #pragma unroll
  for (int h = 0; h < cH; ++h) y[h] = x[h] * sc;
}

// per-column sum / sumsq via atomics (small tensors only)
__global__ void k_colstats(const float* __restrict__ X, int M, int C,
                           float* __restrict__ ssum, float* __restrict__ ssq) {
  int i = blockIdx.x * blockDim.x + threadIdx.x;
  if (i >= M * C) return;
  float v = X[i];
  int c = i % C;
  atomicAdd(&ssum[c], v);
  atomicAdd(&ssq[c], v * v);
}

// in-place: sum,sq -> BN scale/shift folded with affine params:
//   scale = gamma * rsqrt(var+eps); shift = beta - mean*scale
__global__ void k_finalize(float* __restrict__ ssum, float* __restrict__ ssq, int C,
                           float inv_cnt, const float* __restrict__ gamma,
                           const float* __restrict__ beta) {
  int c = blockIdx.x * blockDim.x + threadIdx.x;
  if (c >= C) return;
  float m  = ssum[c] * inv_cnt;
  float va = ssq[c] * inv_cnt - m * m;
  float sc = gamma[c] * rsqrtf(va + 1e-5f);
  ssum[c] = sc;
  ssq[c]  = beta[c] - m * sc;
}

// Y = X*scale + shift (optional relu)
__global__ void k_bn_apply(const float* __restrict__ X, float* __restrict__ Y,
                           const float* __restrict__ scale, const float* __restrict__ shift,
                           int M, int C, int dorelu) {
  int i = blockIdx.x * blockDim.x + threadIdx.x;
  if (i >= M * C) return;
  int c = i % C;
  float v = X[i] * scale[c] + shift[c];
  if (dorelu) v = fmaxf(v, 0.f);
  Y[i] = v;
}

// ---------------- sparse gcn-norm SpMM ----------------
__global__ void k_edge_deg(const int* __restrict__ dst, const float* __restrict__ w,
                           float* __restrict__ deg, int E) {
  int e = blockIdx.x * blockDim.x + threadIdx.x;
  if (e < E) atomicAdd(&deg[dst[e]], w[e]);
}

__global__ void k_rsqrt_inplace(float* __restrict__ d, int n) {
  int i = blockIdx.x * blockDim.x + threadIdx.x;
  if (i >= n) return;
  float v = d[i];
  d[i] = (v > 0.f) ? rsqrtf(v) : 0.f;
}

// self-loop contribution: agg[n,h] = dinv[n]^2 * X[(srcmod? n%srcmod : n), h]
__global__ void k_agg_init(const float* __restrict__ X, const float* __restrict__ dinv,
                           float* __restrict__ agg, int n, int srcmod) {
  int i = blockIdx.x * blockDim.x + threadIdx.x;
  if (i >= n * cH) return;
  int node = i >> 6, h = i & 63;
  int xr = srcmod ? (node % srcmod) : node;
  float di = dinv[node];
  agg[i] = di * di * X[(size_t)xr * cH + h];
}

__global__ void k_edge_scatter(const int* __restrict__ src, const int* __restrict__ dst,
                               const float* __restrict__ w, const float* __restrict__ dinv,
                               const float* __restrict__ X, float* __restrict__ agg,
                               int E, int srcmod) {
  int e = blockIdx.x * blockDim.x + threadIdx.x;
  if (e >= E) return;
  int s = src[e], d = dst[e];
  float coef = dinv[s] * w[e] * dinv[d];
  const float* xp = X + (size_t)(srcmod ? (s % srcmod) : s) * cH;
  float* ap = agg + (size_t)d * cH;
  #pragma unroll 8
  for (int h = 0; h < cH; ++h) atomicAdd(&ap[h], coef * xp[h]);
}

// ---------------- WMMA GEMM: C[M,Nc] = xform(A[M,K]) @ W[Nc,K]^T + bias ----------------
// XMODE: 0 = raw A, 2 = BN(scale/shift)+ReLU applied to A while loading.
// Out-of-range A rows are CLAMPED to M-1 (valid memory, result rows masked at store)
// so fragment loads are unconditional -> vector b128 loads, no EXEC juggling.
template <int XMODE, int KK>
__global__ void wmma_gemm_bt(const float* __restrict__ A, const float* __restrict__ W,
                             const float* __restrict__ bias, float* __restrict__ C,
                             int M, int Nc,
                             const float* __restrict__ xscale,
                             const float* __restrict__ xshift,
                             int accum,
                             float* __restrict__ ssum, float* __restrict__ ssq) {
  const int lane = threadIdx.x & 31;
  const int wave = threadIdx.x >> 5;
  const int m0 = (blockIdx.x * 8 + wave) * 16;
  if (m0 >= M) return;                       // wave-uniform: EXEC stays all-ones
  const int n0 = blockIdx.y * 16;
  const int r  = lane & 15;
  const int kbA = (lane < 16) ? 0 : 8;       // A frag K base (ISA 16-bit A 16x32 layout)
  const int kbB = (lane < 16) ? 0 : 8;       // B frag K base (same table order as A)

  int arow = m0 + r;
  if (arow >= M) arow = M - 1;               // clamp: D row m depends only on A row m
  const float* ap = A + (size_t)arow * KK;
  const float* wp = W + (size_t)(n0 + r) * KK;

  v8f acc = {};
  #pragma unroll
  for (int k0 = 0; k0 < KK; k0 += 32) {
    if (k0 + 32 < KK) __builtin_prefetch(ap + k0 + 32, 0, 0);
    v16bf af, bf;
    #pragma unroll
    for (int j = 0; j < 8; ++j) {
      int c0 = k0 + kbA + j;
      int c1 = c0 + 16;
      float a0 = ap[c0];
      float a1 = ap[c1];
      if (XMODE >= 1) {
        a0 = a0 * xscale[c0] + xshift[c0];
        a1 = a1 * xscale[c1] + xshift[c1];
      }
      if (XMODE == 2) { a0 = fmaxf(a0, 0.f); a1 = fmaxf(a1, 0.f); }
      af[j]     = (__bf16)a0;
      af[8 + j] = (__bf16)a1;
    }
    #pragma unroll
    for (int j = 0; j < 16; ++j) bf[j] = (__bf16)wp[k0 + kbB + (j & 7) + ((j >> 3) << 4)];
    acc = __builtin_amdgcn_wmma_f32_16x16x32_bf16(false, af, false, bf,
                                                  (short)0, acc, false, false);
  }

  const int col = n0 + r;
  const int rowbase = m0 + ((lane < 16) ? 0 : 8);
  const float bv = bias ? bias[col] : 0.f;
  float s = 0.f, sq = 0.f;
  if (m0 + 16 <= M) {                        // full tile: no per-row guards
    #pragma unroll
    for (int v = 0; v < 8; ++v) {
      float val = acc[v] + bv;
      size_t idx = (size_t)(rowbase + v) * Nc + col;
      if (accum) val += C[idx];
      C[idx] = val;
      s += val; sq += val * val;
    }
  } else {
    #pragma unroll
    for (int v = 0; v < 8; ++v) {
      int row = rowbase + v;
      if (row < M) {
        float val = acc[v] + bv;
        size_t idx = (size_t)row * Nc + col;
        if (accum) val += C[idx];
        C[idx] = val;
        s += val; sq += val * val;
      }
    }
  }
  if (ssum) { atomicAdd(&ssum[col], s); atomicAdd(&ssq[col], sq); }
}

// ---------------- fusion / decoder kernels ----------------
__global__ void k_base1(const float* __restrict__ base0, const float* __restrict__ poslin,
                        float* __restrict__ out) {
  int i = blockIdx.x * blockDim.x + threadIdx.x;
  if (i >= cN * cH) return;
  int n = i >> 6, h = i & 63;
  int g = n % cG;
  out[i] = base0[(size_t)g * cH + h] + 0.2f * poslin[i];
}

// Y = X*scale+shift + emb_total[b], accumulate column stats of Y
__global__ void k_bn_add_stats(const float* __restrict__ X,
                               const float* __restrict__ scale, const float* __restrict__ shift,
                               const float* __restrict__ et, float* __restrict__ Y,
                               float* __restrict__ ssum, float* __restrict__ ssq) {
  int h = threadIdx.x & 63;
  int rowStart  = blockIdx.x * (blockDim.x >> 6) + (threadIdx.x >> 6);
  int rowStride = gridDim.x * (blockDim.x >> 6);
  float s = 0.f, sq = 0.f;
  float sc = scale[h], sh = shift[h];
  for (int n = rowStart; n < cN; n += rowStride) {
    int b = n / cG;
    float v = X[(size_t)n * cH + h] * sc + sh + et[(size_t)b * cH + h];
    Y[(size_t)n * cH + h] = v;
    s += v; sq += v * v;
  }
  atomicAdd(&ssum[h], s);
  atomicAdd(&ssq[h], sq);
}

__global__ void k_gather_bn(const float* __restrict__ X, const int* __restrict__ idx,
                            const float* __restrict__ scale, const float* __restrict__ shift,
                            float* __restrict__ Y) {
  int i = blockIdx.x * blockDim.x + threadIdx.x;
  if (i >= cB * cH) return;
  int b = i >> 6, h = i & 63;
  int p = idx[b];
  Y[i] = X[(size_t)p * cH + h] * scale[h] + shift[h];
}

// w1[b,g] = dot(BN(out[n,:]), indv_w1[g,:]) + indv_b1[g]
__global__ void k_w1(const float* __restrict__ X,
                     const float* __restrict__ scale, const float* __restrict__ shift,
                     const float* __restrict__ iw1, const float* __restrict__ ib1,
                     float* __restrict__ w1) {
  int n = blockIdx.x * blockDim.x + threadIdx.x;
  if (n >= cN) return;
  int g = n % cG;
  const float* o = X + (size_t)n * cH;
  const float* wv = iw1 + (size_t)g * cH;
  float acc = 0.f;
  #pragma unroll
  for (int h = 0; h < cH; ++h) acc += (o[h] * scale[h] + shift[h]) * wv[h];
  w1[n] = acc + ib1[g];
}

// cgpre[b,h] = sum_g w1[b,g]*cg_W1[h,g] + cg_b1[h]   (one block per (b,h))
__global__ void k_cgpre(const float* __restrict__ w1, const float* __restrict__ cgW1,
                        const float* __restrict__ cgb1, float* __restrict__ cgpre) {
  int b = blockIdx.x >> 6;
  int h = blockIdx.x & 63;
  float s = 0.f;
  for (int g = threadIdx.x; g < cG; g += blockDim.x)
    s += w1[(size_t)b * cG + g] * cgW1[(size_t)h * cG + g];
  __shared__ float red[256];
  red[threadIdx.x] = s;
  __syncthreads();
  for (int st = 128; st > 0; st >>= 1) {
    if ((int)threadIdx.x < st) red[threadIdx.x] += red[threadIdx.x + st];
    __syncthreads();
  }
  if (threadIdx.x == 0) cgpre[(size_t)b * cH + h] = red[0] + cgb1[h];
}

// BN over 32 rows (one block of 64 threads, one channel per thread)
__global__ void k_bn32(const float* __restrict__ X, const float* __restrict__ gamma,
                       const float* __restrict__ beta, float* __restrict__ Y, int dorelu) {
  int h = threadIdx.x;
  if (h >= cH) return;
  float s = 0.f, sq = 0.f;
  for (int b = 0; b < cB; ++b) { float v = X[(size_t)b * cH + h]; s += v; sq += v * v; }
  float m = s / cB;
  float var = sq / cB - m * m;
  float rs = rsqrtf(var + 1e-5f);
  for (int b = 0; b < cB; ++b) {
    float v = (X[(size_t)b * cH + h] - m) * rs * gamma[h] + beta[h];
    if (dorelu) v = fmaxf(v, 0.f);
    Y[(size_t)b * cH + h] = v;
  }
}

__global__ void k_mm64(const float* __restrict__ X, const float* __restrict__ W,
                       const float* __restrict__ bias, float* __restrict__ Y) {
  int i = blockIdx.x * blockDim.x + threadIdx.x;
  if (i >= cB * cH) return;
  int b = i >> 6, h = i & 63;
  float s = bias[h];
  #pragma unroll
  for (int k = 0; k < cH; ++k) s += X[(size_t)b * cH + k] * W[(size_t)h * cH + k];
  Y[i] = s;
}

__global__ void k_final(const float* __restrict__ w1, const float* __restrict__ cgf,
                        const float* __restrict__ iw2, const float* __restrict__ ib2,
                        const float* __restrict__ x, float* __restrict__ out) {
  int n = blockIdx.x * blockDim.x + threadIdx.x;
  if (n >= cN) return;
  int b = n / cG, g = n % cG;
  const float* r = iw2 + (size_t)g * (cH + 1);
  const float* c = cgf + (size_t)b * cH;
  float s = w1[n] * r[0] + ib2[g] + x[n];
  #pragma unroll
  for (int h = 0; h < cH; ++h) s += c[h] * r[1 + h];
  out[n] = s;
}

// ---------------- host orchestration ----------------
extern "C" void kernel_launch(void* const* d_in, const int* in_sizes, int n_in,
                              void* d_out, int out_size, void* d_ws, size_t ws_size,
                              hipStream_t stream) {
  const float* x        = (const float*)d_in[0];
  const int*   pert_idx = (const int*)  d_in[1];
  const int*   ei_co    = (const int*)  d_in[2];
  const float* w_co     = (const float*)d_in[3];
  const int*   ei_go    = (const int*)  d_in[4];
  const float* w_go     = (const float*)d_in[5];
  const int*   ei_bg    = (const int*)  d_in[6];
  const float* w_bg     = (const float*)d_in[7];
  const float* gene_tab = (const float*)d_in[8];
  const float* pos_tab  = (const float*)d_in[9];
  const float* go_tab   = (const float*)d_in[10];
  const float* bg_tab   = (const float*)d_in[11];
  const float* bn_g     = (const float*)d_in[12];
  const float* bn_b     = (const float*)d_in[13];
  const float* sg_W     = (const float*)d_in[14];
  const float* sg_b     = (const float*)d_in[15];
  const float* mlp_W    = (const float*)d_in[16];
  const float* mlp_b    = (const float*)d_in[17];
  const float* mlp_g    = (const float*)d_in[18];
  const float* mlp_bt   = (const float*)d_in[19];
  const float* rec_W1   = (const float*)d_in[20];
  const float* rec_b1   = (const float*)d_in[21];
  const float* rec_g1   = (const float*)d_in[22];
  const float* rec_bt1  = (const float*)d_in[23];
  const float* rec_W2   = (const float*)d_in[24];
  const float* rec_b2   = (const float*)d_in[25];
  const float* rec_g2   = (const float*)d_in[26];
  const float* rec_bt2  = (const float*)d_in[27];
  const float* indv_w1  = (const float*)d_in[28];
  const float* indv_b1  = (const float*)d_in[29];
  const float* cg_W1    = (const float*)d_in[30];
  const float* cg_b1    = (const float*)d_in[31];
  const float* cg_g1    = (const float*)d_in[32];
  const float* cg_bt1   = (const float*)d_in[33];
  const float* cg_W2    = (const float*)d_in[34];
  const float* cg_b2    = (const float*)d_in[35];
  const float* cg_g2    = (const float*)d_in[36];
  const float* cg_bt2   = (const float*)d_in[37];
  const float* indv_w2  = (const float*)d_in[38];
  const float* indv_b2  = (const float*)d_in[39];

  // ---- workspace carve-up (floats) ----
  float* ws = (float*)d_ws;
  size_t off = 0;
  auto alloc = [&](size_t n) { float* p = ws + off; off += n; return p; };
  float* bufA   = alloc((size_t)cN * 128);        // rec1 hidden [N,128]
  float* bufB   = alloc((size_t)cN * cH);
  float* bufC   = alloc((size_t)cN * cH);
  float* bufD   = bufA + (size_t)cN * cH;         // alias: dead before bufA is written
  float* g_rn   = alloc((size_t)cG * cH);
  float* pos_rn = alloc((size_t)cG * cH);
  float* base0  = alloc((size_t)cG * cH);
  float* go_rn  = alloc((size_t)cP * cH);
  float* bg_rn  = alloc((size_t)cP * cH);
  float* pagg   = alloc((size_t)cP * cH);
  float* plin   = alloc((size_t)cP * cH);
  float* pt1    = alloc((size_t)cP * cH);
  float* pt2    = alloc((size_t)cP * cH);
  float* deg_co = alloc(cN);
  float* deg_p  = alloc(cP);
  float* w1buf  = alloc(cN);
  float* pe     = alloc(cB * cH);
  float* tA     = alloc(cB * cH);
  float* tB     = alloc(cB * cH);
  float* etot   = alloc(cB * cH);
  float* cgpre  = alloc(cB * cH);
  float* cgh    = alloc(cB * cH);
  float* cg2    = alloc(cB * cH);
  float* cgf    = alloc(cB * cH);
  float* stats  = alloc(10 * 256);                // slot i: scale/sum=+0, shift/sq=+128
  auto ST = [&](int i) { return stats + (size_t)i * 256; };
  // slots: 0 emb | 1 mlp1a | 2 mlp1b | 3 bn_pert_base | 4 rec1(128) | 5 rec2
  //        6 mlp0a | 7 mlp0b | 8 mlp2a | 9 mlp2b

  auto cdiv = [](int a, int b) { return (a + b - 1) / b; };
  auto fin = [&](float* st, int C, float inv, const float* gm, const float* bt) {
    k_finalize<<<1, 128, 0, stream>>>(st, st + 128, C, inv, gm, bt);
  };
  // xmode: 0 raw, 2 BN+ReLU folded into A load
  auto gemm = [&](const float* A, const float* W, const float* bias, float* C,
                  int M, int K, int Nc, const float* st_in, int xmode,
                  int accum, float* st_out) {
    dim3 grid(cdiv(M, 128), Nc / 16);
    const float* xs = st_in;
    const float* xh = st_in ? st_in + 128 : nullptr;
    float* ss = st_out;
    float* sq = st_out ? st_out + 128 : nullptr;
    if (K == 64) {
      if (xmode == 0)
        wmma_gemm_bt<0, 64><<<grid, 256, 0, stream>>>(A, W, bias, C, M, Nc, xs, xh, accum, ss, sq);
      else
        wmma_gemm_bt<2, 64><<<grid, 256, 0, stream>>>(A, W, bias, C, M, Nc, xs, xh, accum, ss, sq);
    } else {
      wmma_gemm_bt<2, 128><<<grid, 256, 0, stream>>>(A, W, bias, C, M, Nc, xs, xh, accum, ss, sq);
    }
  };

  // zero stats
  k_fill<<<cdiv(10 * 256, 256), 256, 0, stream>>>(stats, 0.f, 10 * 256);

  // renorm embedding tables
  k_renorm<<<cdiv(cG, 256), 256, 0, stream>>>(gene_tab, g_rn, cG);
  k_renorm<<<cdiv(cG, 256), 256, 0, stream>>>(pos_tab, pos_rn, cG);
  k_renorm<<<cdiv(cP, 256), 256, 0, stream>>>(go_tab, go_rn, cP);
  k_renorm<<<cdiv(cP, 256), 256, 0, stream>>>(bg_tab, bg_rn, cP);

  // base0 = relu(bn(emb))  (tiled stats == per-G stats)
  k_colstats<<<cdiv(cG * cH, 256), 256, 0, stream>>>(g_rn, cG, cH, ST(0), ST(0) + 128);
  fin(ST(0), cH, 1.f / cG, bn_g, bn_b);
  k_bn_apply<<<cdiv(cG * cH, 256), 256, 0, stream>>>(g_rn, base0, ST(0), ST(0) + 128,
                                                     cG, cH, 1);

  // --- SGConv on co-expression graph (N nodes) ---
  const int* co_src = ei_co, *co_dst = ei_co + cECO;
  k_fill<<<cdiv(cN, 256), 256, 0, stream>>>(deg_co, 1.f, cN);
  k_edge_deg<<<cdiv(cECO, 256), 256, 0, stream>>>(co_dst, w_co, deg_co, cECO);
  k_rsqrt_inplace<<<cdiv(cN, 256), 256, 0, stream>>>(deg_co, cN);
  k_agg_init<<<cdiv(cN * cH, 256), 256, 0, stream>>>(pos_rn, deg_co, bufB, cN, cG);
  k_edge_scatter<<<cdiv(cECO, 256), 256, 0, stream>>>(co_src, co_dst, w_co, deg_co,
                                                      pos_rn, bufB, cECO, cG);
  gemm(bufB, sg_W, sg_b, bufC, cN, cH, cH, nullptr, 0, 0, nullptr);

  // base1 = base0[tiled] + 0.2*pos
  k_base1<<<cdiv(cN * cH, 256), 256, 0, stream>>>(base0, bufC, bufB);

  // mlp#1 (emb_trans_v2) GEMM1 with stats
  gemm(bufB, mlp_W + 2 * 4096, mlp_b + 2 * 64, bufC, cN, cH, cH, nullptr, 0, 0, ST(1));
  fin(ST(1), cH, 1.f / cN, mlp_g + 2 * 64, mlp_bt + 2 * 64);

  // --- perturbation branch ---
  const int* go_src = ei_go, *go_dst = ei_go + cEGG;
  k_fill<<<cdiv(cP, 256), 256, 0, stream>>>(deg_p, 1.f, cP);
  k_edge_deg<<<cdiv(cEGG, 256), 256, 0, stream>>>(go_dst, w_go, deg_p, cEGG);
  k_rsqrt_inplace<<<cdiv(cP, 256), 256, 0, stream>>>(deg_p, cP);
  k_agg_init<<<cdiv(cP * cH, 256), 256, 0, stream>>>(go_rn, deg_p, pagg, cP, 0);
  k_edge_scatter<<<cdiv(cEGG, 256), 256, 0, stream>>>(go_src, go_dst, w_go, deg_p,
                                                      go_rn, pagg, cEGG, 0);
  gemm(pagg, sg_W + 4096, sg_b + 64, plin, cP, cH, cH, nullptr, 0, 0, nullptr);

  const int* bg_src = ei_bg, *bg_dst = ei_bg + cEGG;
  k_fill<<<cdiv(cP, 256), 256, 0, stream>>>(deg_p, 1.f, cP);
  k_edge_deg<<<cdiv(cEGG, 256), 256, 0, stream>>>(bg_dst, w_bg, deg_p, cEGG);
  k_rsqrt_inplace<<<cdiv(cP, 256), 256, 0, stream>>>(deg_p, cP);
  k_agg_init<<<cdiv(cP * cH, 256), 256, 0, stream>>>(bg_rn, deg_p, pagg, cP, 0);
  k_edge_scatter<<<cdiv(cEGG, 256), 256, 0, stream>>>(bg_src, bg_dst, w_bg, deg_p,
                                                      bg_rn, pagg, cEGG, 0);
  gemm(pagg, sg_W + 2 * 4096, sg_b + 2 * 64, plin, cP, cH, cH, nullptr, 0, 1, nullptr);

  // mlp#0 over P rows
  gemm(plin, mlp_W, mlp_b, pt1, cP, cH, cH, nullptr, 0, 0, ST(6));
  fin(ST(6), cH, 1.f / cP, mlp_g, mlp_bt);
  gemm(pt1, mlp_W + 4096, mlp_b + 64, pt2, cP, cH, cH, ST(6), 2, 0, ST(7));
  fin(ST(7), cH, 1.f / cP, mlp_g + 64, mlp_bt + 64);
  k_gather_bn<<<cdiv(cB * cH, 256), 256, 0, stream>>>(pt2, pert_idx, ST(7), ST(7) + 128, pe);

  // mlp#2 over B=32 rows
  gemm(pe, mlp_W + 4 * 4096, mlp_b + 4 * 64, tA, cB, cH, cH, nullptr, 0, 0, ST(8));
  fin(ST(8), cH, 1.f / cB, mlp_g + 4 * 64, mlp_bt + 4 * 64);
  gemm(tA, mlp_W + 5 * 4096, mlp_b + 5 * 64, tB, cB, cH, cH, ST(8), 2, 0, ST(9));
  fin(ST(9), cH, 1.f / cB, mlp_g + 5 * 64, mlp_bt + 5 * 64);
  k_bn_apply<<<cdiv(cB * cH, 256), 256, 0, stream>>>(tB, etot, ST(9), ST(9) + 128, cB, cH, 0);

  // mlp#1 GEMM2 (reads bufC with BN+ReLU of layer1 stats)
  gemm(bufC, mlp_W + 3 * 4096, mlp_b + 3 * 64, bufD, cN, cH, cH, ST(1), 2, 0, ST(2));
  fin(ST(2), cH, 1.f / cN, mlp_g + 3 * 64, mlp_bt + 3 * 64);

  // base = BN(mlp_out) + emb_total, with stats for bn_pert_base
  k_bn_add_stats<<<1024, 256, 0, stream>>>(bufD, ST(2), ST(2) + 128, etot, bufB,
                                           ST(3), ST(3) + 128);
  fin(ST(3), cH, 1.f / cN, bn_g + 64, bn_b + 64);

  // recovery MLP: [N,64]->[N,128]->[N,64]
  gemm(bufB, rec_W1, rec_b1, bufA, cN, cH, 128, ST(3), 2, 0, ST(4));
  fin(ST(4), 128, 1.f / cN, rec_g1, rec_bt1);
  gemm(bufA, rec_W2, rec_b2, bufC, cN, 128, cH, ST(4), 2, 0, ST(5));
  fin(ST(5), cH, 1.f / cN, rec_g2, rec_bt2);

  // per-gene decoder 1 (folds final BN of rec2)
  k_w1<<<cdiv(cN, 256), 256, 0, stream>>>(bufC, ST(5), ST(5) + 128, indv_w1, indv_b1, w1buf);

  // cross-gene state
  k_cgpre<<<cB * cH, 256, 0, stream>>>(w1buf, cg_W1, cg_b1, cgpre);
  k_bn32<<<1, 64, 0, stream>>>(cgpre, cg_g1, cg_bt1, cgh, 1);
  k_mm64<<<cdiv(cB * cH, 256), 256, 0, stream>>>(cgh, cg_W2, cg_b2, cg2);
  k_bn32<<<1, 64, 0, stream>>>(cg2, cg_g2, cg_bt2, cgf, 0);

  // per-gene decoder 2 + residual
  k_final<<<cdiv(cN, 256), 256, 0, stream>>>(w1buf, cgf, indv_w2, indv_b2, x, (float*)d_out);
}
